// PureAttention_50448685859256
// MI455X (gfx1250) — matrix operand order repaired and checked by reading, not compiled, not run on hardware
//
#include <hip/hip_runtime.h>

#define B_  2
#define S_  2048
#define D_  2048
#define H_  16
#define HD_ 128
#define QT  128      // queries per workgroup (8 waves x 16 rows)
#define KT  32       // keys per inner iteration
#define NW  8        // waves per workgroup
#define SCALE 0.08838834764831845f   // 1/sqrt(128)

// padded LDS row strides (elements) to break mod-64-bank alignment
#define HDP 136      // Kt row stride: 272 B  (2-row lane stride = 8 banks -> 2-way optimal)
#define KTP 48       // Vt row stride:  96 B  (24-bank lane stride -> 2-way optimal)
#define PLP 40       // Pl row stride:  80 B  (20-bank lane stride -> conflict-free 16B reads)

typedef __attribute__((ext_vector_type(16))) __bf16 v16bf;
typedef __attribute__((ext_vector_type(8)))  __bf16 v8bf;
typedef __attribute__((ext_vector_type(2)))  __bf16 v2bf;
typedef __attribute__((ext_vector_type(8)))  float  v8f;

// ---- 16-lane butterfly reductions via DPP row ops (no LDS-pipe traffic) ----
#if __has_builtin(__builtin_amdgcn_update_dpp)
template <int CTRL>
__device__ __forceinline__ float dpp_mov(float v) {
    return __int_as_float(__builtin_amdgcn_update_dpp(
        0, __float_as_int(v), CTRL, 0xf, 0xf, true));
}
__device__ __forceinline__ float red_max16(float v) {
    v = fmaxf(v, dpp_mov<0xB1>(v));    // quad_perm [1,0,3,2]  (xor 1)
    v = fmaxf(v, dpp_mov<0x4E>(v));    // quad_perm [2,3,0,1]  (xor 2)
    v = fmaxf(v, dpp_mov<0x141>(v));   // row_half_mirror      (covers xor 4)
    v = fmaxf(v, dpp_mov<0x140>(v));   // row_mirror           (covers xor 8)
    return v;
}
__device__ __forceinline__ float red_sum16(float v) {
    v += dpp_mov<0xB1>(v);
    v += dpp_mov<0x4E>(v);
    v += dpp_mov<0x141>(v);
    v += dpp_mov<0x140>(v);
    return v;
}
#else
__device__ __forceinline__ float red_max16(float v) {
    #pragma unroll
    for (int o = 8; o >= 1; o >>= 1) v = fmaxf(v, __shfl_xor(v, o, 16));
    return v;
}
__device__ __forceinline__ float red_sum16(float v) {
    #pragma unroll
    for (int o = 8; o >= 1; o >>= 1) v += __shfl_xor(v, o, 16);
    return v;
}
#endif

// load 16 contiguous bf16 from LDS as a B-fragment (two 16B halves)
__device__ __forceinline__ v16bf ld_bfrag(const __bf16* p) {
    const v8bf lo = *reinterpret_cast<const v8bf*>(p);
    const v8bf hi = *reinterpret_cast<const v8bf*>(p + 8);
    v16bf r;
    #pragma unroll
    for (int e = 0; e < 8; ++e) { r[e] = lo[e]; r[8 + e] = hi[e]; }
    return r;
}

__global__ __launch_bounds__(256)
void attn_fwd_kernel(const float* __restrict__ Qg, const float* __restrict__ Kg,
                     const float* __restrict__ Vg, const float* __restrict__ Mask,
                     float* __restrict__ Out, float* __restrict__ AW)
{
    __shared__ alignas(32) __bf16 Kt[KT * HDP];        // [key][dim]  padded
    __shared__ alignas(32) __bf16 Vt[HD_ * KTP];       // [dim][key]  padded (T)
    __shared__ alignas(32) __bf16 Pl[NW][16 * PLP];    // per-wave P  padded
    __shared__ float Ml[S_];                           // mask row

    const int tid  = threadIdx.x;
    const int wave = tid >> 5;
    const int lane = tid & 31;
    const int n16  = lane & 15;    // N (col) index inside a 16-wide tile
    const int half = lane >> 4;    // K-range / M-offset selector per ISA layout

    const int qb = blockIdx.x & 15;          // S/QT = 16 query blocks
    const int h  = (blockIdx.x >> 4) & 15;   // head
    const int b  = blockIdx.x >> 8;          // batch

    const int q0 = qb * QT + wave * 16;      // this wave's first query row

    // ---- stage additive mask row for this batch into LDS ----
    for (int i = tid; i < S_; i += 256) Ml[i] = Mask[b * S_ + i];

    // ---- load this wave's 16x128 Q strip as 4 bf16 A-fragments (16x32 each) ----
    v16bf qa[4];
    {
        const int qg = q0 + n16;                                  // A layout: M = lane%16
        const float* qp = Qg + ((long)(b * S_ + qg)) * D_ + h * HD_;
        #pragma unroll
        for (int c = 0; c < 4; ++c) {
            const float* qlo = qp + c * 32 + 16 * half;
            const float* qhi = qp + c * 32 + 16 * half + 8;
            #pragma unroll
            for (int e = 0; e < 8; ++e) {
                qa[c][e]     = (__bf16)qlo[e];
                qa[c][8 + e] = (__bf16)qhi[e];
            }
        }
    }

    // online softmax stats: rows M = j + 8*half (C-matrix layout)
    float m[8], l[8];
    #pragma unroll
    for (int j = 0; j < 8; ++j) { m[j] = -3.0e38f; l[j] = 0.0f; }

    // key-subtile interleave: subtile0 covers keys {2n}, subtile1 keys {2n+1}
    const int kr0 = 2 * n16;        // Kt row for subtile 0
    const int kr1 = 2 * n16 + 1;    // Kt row for subtile 1

    // ================= pass 1: score streaming, running max/sum =================
    for (int kb = 0; kb < S_ / KT; ++kb) {
        __syncthreads();
        {   // stage 32x128 K tile (f32 -> bf16)
            int r  = tid >> 3;                 // key row 0..31
            int d0 = (tid & 7) * 16;           // 16 dims per thread
            const float* kp = Kg + ((long)(b * S_ + kb * KT + r)) * D_ + h * HD_ + d0;
            __builtin_prefetch(kp + (long)KT * D_, 0, 1);   // global_prefetch_b8
            #pragma unroll
            for (int e = 0; e < 16; ++e) Kt[r * HDP + d0 + e] = (__bf16)kp[e];
        }
        __syncthreads();

        v8f c0 = {}; v8f c1 = {};
        #pragma unroll
        for (int c = 0; c < 4; ++c) {
            const v16bf b0 = ld_bfrag(&Kt[kr0 * HDP + c * 32 + 16 * half]);
            const v16bf b1 = ld_bfrag(&Kt[kr1 * HDP + c * 32 + 16 * half]);
            c0 = __builtin_amdgcn_wmma_f32_16x16x32_bf16(false, qa[c], false, b0, (short)0, c0, false, false);
            c1 = __builtin_amdgcn_wmma_f32_16x16x32_bf16(false, qa[c], false, b1, (short)0, c1, false, false);
        }
        const float2 mk = *reinterpret_cast<const float2*>(&Ml[kb * KT + 2 * n16]);
        #pragma unroll
        for (int j = 0; j < 8; ++j) {
            float s0 = c0[j] * SCALE + mk.x;
            float s1 = c1[j] * SCALE + mk.y;
            float rmax = red_max16(fmaxf(s0, s1));
            float mn = fmaxf(m[j], rmax);
            float se = red_sum16(__expf(s0 - mn) + __expf(s1 - mn));
            l[j] = l[j] * __expf(m[j] - mn) + se;
            m[j] = mn;
        }
    }

    // fold 1/l into the exponent: p = exp(s - (m + log l))
    float ms[8];
    #pragma unroll
    for (int j = 0; j < 8; ++j) ms[j] = m[j] + __logf(l[j]);

    const v8f vzero = {};
    v8f oacc[8];
    #pragma unroll
    for (int t = 0; t < 8; ++t) oacc[t] = vzero;

    // ====== pass 2: recompute scores, write normalized weights, O += P*V ======
    for (int kb = 0; kb < S_ / KT; ++kb) {
        __syncthreads();
        {   // stage K tile and transposed V tile
            int r  = tid >> 3;
            int d0 = (tid & 7) * 16;
            const float* kp = Kg + ((long)(b * S_ + kb * KT + r)) * D_ + h * HD_ + d0;
            const float* vp = Vg + ((long)(b * S_ + kb * KT + r)) * D_ + h * HD_ + d0;
            __builtin_prefetch(vp + (long)KT * D_, 0, 1);
            #pragma unroll
            for (int e = 0; e < 16; ++e) {
                Kt[r * HDP + d0 + e]   = (__bf16)kp[e];
                Vt[(d0 + e) * KTP + r] = (__bf16)vp[e];    // transpose: [dim][key]
            }
        }
        __syncthreads();

        v8f c0 = {}; v8f c1 = {};
        #pragma unroll
        for (int c = 0; c < 4; ++c) {
            const v16bf b0 = ld_bfrag(&Kt[kr0 * HDP + c * 32 + 16 * half]);
            const v16bf b1 = ld_bfrag(&Kt[kr1 * HDP + c * 32 + 16 * half]);
            c0 = __builtin_amdgcn_wmma_f32_16x16x32_bf16(false, qa[c], false, b0, (short)0, c0, false, false);
            c1 = __builtin_amdgcn_wmma_f32_16x16x32_bf16(false, qa[c], false, b1, (short)0, c1, false, false);
        }
        const float2 mk = *reinterpret_cast<const float2*>(&Ml[kb * KT + 2 * n16]);
        #pragma unroll
        for (int j = 0; j < 8; ++j) {
            int row  = j + 8 * half;                       // C layout row
            float p0 = __expf(c0[j] * SCALE + mk.x - ms[j]);
            float p1 = __expf(c1[j] * SCALE + mk.y - ms[j]);
            long qg    = q0 + row;
            long awoff = (((long)(b * H_ + h)) * S_ + qg) * S_ + (long)kb * KT;
            float2 pv; pv.x = p0; pv.y = p1;               // adjacent keys 2n, 2n+1
            *reinterpret_cast<float2*>(&AW[awoff + 2 * n16]) = pv;   // one b64 store
            v2bf pp; pp[0] = (__bf16)p0; pp[1] = (__bf16)p1;
            *reinterpret_cast<v2bf*>(&Pl[wave][row * PLP + 2 * n16]) = pp;  // one b32 store
        }
        // Pl is per-wave private; same-wave LDS ops are in-order -> no barrier needed

        // rebuild P as a 16x32 bf16 A-fragment: keys 16*half .. 16*half+15
        const v8bf plo = *reinterpret_cast<const v8bf*>(&Pl[wave][n16 * PLP + 16 * half]);
        const v8bf phi = *reinterpret_cast<const v8bf*>(&Pl[wave][n16 * PLP + 16 * half + 8]);
        v16bf pa;
        #pragma unroll
        for (int e = 0; e < 8; ++e) { pa[e] = plo[e]; pa[8 + e] = phi[e]; }

        // O[16q x 128d] += P[16x32] * V[32x128] : 8 d-subtiles
        #pragma unroll
        for (int t = 0; t < 8; ++t) {
            const v16bf vb = ld_bfrag(&Vt[(t * 16 + n16) * KTP + 16 * half]);
            oacc[t] = __builtin_amdgcn_wmma_f32_16x16x32_bf16(false, pa, false, vb, (short)0, oacc[t], false, false);
        }
    }

    // ---- write attn_output ----
    #pragma unroll
    for (int t = 0; t < 8; ++t) {
        #pragma unroll
        for (int j = 0; j < 8; ++j) {
            int row = j + 8 * half;
            long qg = q0 + row;
            Out[((long)(b * S_ + qg)) * D_ + h * HD_ + t * 16 + n16] = oacc[t][j];
        }
    }
}

extern "C" void kernel_launch(void* const* d_in, const int* in_sizes, int n_in,
                              void* d_out, int out_size, void* d_ws, size_t ws_size,
                              hipStream_t stream) {
    const float* Q = (const float*)d_in[0];
    const float* K = (const float*)d_in[1];
    const float* V = (const float*)d_in[2];
    const float* M = (const float*)d_in[3];
    float* out = (float*)d_out;                       // attn_output first
    float* aw  = out + (long)B_ * S_ * D_;            // then attn_weights
    dim3 grid(B_ * H_ * (S_ / QT));                   // 512 workgroups
    attn_fwd_kernel<<<grid, 256, 0, stream>>>(Q, K, V, M, out, aw);
}
